// OneShotStereoNet_44083544326887
// MI455X (gfx1250) — matrix operand order repaired and checked by reading, not compiled, not run on hardware
//
#include <hip/hip_runtime.h>
#include <hip/hip_bf16.h>
#include <math.h>

// ---------------------------------------------------------------------------
// MI455X (gfx1250) implementation of the one-shot DiT forward pass.
// All large GEMMs go through v_wmma_f32_16x16x32_bf16 (wave32), with f32
// operands converted to bf16 while staging tiles into LDS. Accumulation is
// f32. The HBM-heavy weight stream (B operand) is double-buffered into LDS
// with GLOBAL_LOAD_ASYNC_TO_LDS (ASYNCcnt) when the toolchain exposes the
// builtins; otherwise it falls back to synchronous staged loads.
// ---------------------------------------------------------------------------

typedef __attribute__((ext_vector_type(16))) __bf16          v16bf;
typedef __attribute__((ext_vector_type(8)))  float           v8f;
typedef __attribute__((ext_vector_type(8)))  unsigned short  v8us;
typedef __attribute__((ext_vector_type(16))) unsigned short  v16us;
typedef __attribute__((ext_vector_type(4)))  int             v4i;

#define HIDDEN   1024
#define N_TOKENS 1024
#define N_HEADS  16
#define HEAD_DIM 64
#define N_LAYERS 10

#define EPI_BIAS  1
#define EPI_GELU  2
#define EPI_RESID 4
#define EPI_MASK  8

#if defined(__gfx1250__) && \
    __has_builtin(__builtin_amdgcn_global_load_async_to_lds_b128) && \
    __has_builtin(__builtin_amdgcn_s_wait_asynccnt)
#define USE_ASYNC_LDS 1
#else
#define USE_ASYNC_LDS 0
#endif

__device__ __forceinline__ unsigned short f2bf(float f) {
  unsigned int u = __float_as_uint(f);
  unsigned int r = u + 0x7FFFu + ((u >> 16) & 1u);   // round-to-nearest-even
  return (unsigned short)(r >> 16);
}

__device__ __forceinline__ float gelu_exact(float x) {
  return 0.5f * x * (1.0f + erff(x * 0.70710678118654752f));
}

// Build a v16bf fragment from two 16-byte LDS chunks (ds_load_b128 x2).
__device__ __forceinline__ v16bf ld_frag(const unsigned short* p0,
                                         const unsigned short* p1) {
  v8us lo = *(const v8us*)p0;
  v8us hi = *(const v8us*)p1;
  v16us t;
#pragma unroll
  for (int e = 0; e < 8; ++e) { t[e] = lo[e]; t[e + 8] = hi[e]; }
  return __builtin_bit_cast(v16bf, t);
}

#if USE_ASYNC_LDS
// One wave instruction: every lane DMAs 16B global -> LDS, tracked by ASYNCcnt.
// Builtin signature (from hipcc diagnostics): (v4i AS(1)*, v4i AS(3)*, imm
// offset, imm cpol).
__device__ __forceinline__ void async_cp16(const float* g, float* l) {
  __builtin_amdgcn_global_load_async_to_lds_b128(
      (__attribute__((address_space(1))) v4i*)g,
      (__attribute__((address_space(3))) v4i*)l, 0, 0);
}
#endif

// ---------------------------------------------------------------------------
// Batched GEMM: C[z] = alpha * A[z](MxK) * B[z](KxN)  (+bias)(gelu)(*rowscale)
//               (+resid), row-major, leading dims lda/ldb/ldc, per-batch
// element strides sA/sB/sC. M multiple of 128, K multiple of 32.
// Block: 128 threads = 4 waves; block tile 128x128; wave tile 64x64
// (4x4 WMMA 16x16 tiles); K-step 32 staged through LDS in bf16.
// ---------------------------------------------------------------------------
#define BM 128
#define BN 128
#define BK 32
#define SA 40   // padded LDS row stride (elements) for A: 80B, 16B aligned
#define SB 40   // padded LDS row stride (elements) for B (stored [n][k])

__global__ __launch_bounds__(128)
void k_gemm(const float* __restrict__ A, const float* __restrict__ B,
            float* C, int M, int N, int K, int lda, int ldb, int ldc,
            long long sA_, long long sB_, long long sC_,
            const float* __restrict__ bias, const float* resid,
            const float* __restrict__ rowscale, float alpha, int flags)
{
  __shared__ __align__(16) unsigned short As[BM * SA];
  __shared__ __align__(16) unsigned short Bs[BN * SB];
#if USE_ASYNC_LDS
  __shared__ __align__(16) float Bfs[2][BK * BN];   // f32 staging, [k][n]
#endif

  const int tid = threadIdx.x;
  const long long bz = blockIdx.z;
  A += bz * sA_;
  B += bz * sB_;
  C += bz * sC_;
  const float* R = resid ? (resid + bz * sC_) : nullptr;

  const int n0 = blockIdx.x * BN;
  const int m0 = blockIdx.y * BM;

  const int wave = tid >> 5;
  const int lane = tid & 31;
  const int lm   = lane & 15;   // row/col within 16x16 tile
  const int lh   = lane >> 4;   // half select (K-group / M-group)
  const int wm0  = (wave & 1) * 64;
  const int wn0  = (wave >> 1) * 64;

  v8f acc[4][4];
  const v8f zf = {0.f, 0.f, 0.f, 0.f, 0.f, 0.f, 0.f, 0.f};
#pragma unroll
  for (int i = 0; i < 4; ++i)
#pragma unroll
    for (int j = 0; j < 4; ++j) acc[i][j] = zf;

  const int a_r  = tid >> 3;          // 0..15
  const int a_kq = (tid & 7) << 2;    // 0..28 step 4
  const int b_kr = tid >> 5;          // 0..3
  const int b_nq = (tid & 31) << 2;   // 0..124 step 4

#if USE_ASYNC_LDS
  const bool nfull = (n0 + BN <= N);
  const int  nk    = K / BK;

  // Stage B K-tile `ti` into Bfs[buf]: async DMA when the whole N-tile is in
  // range, guarded synchronous fill (zero-padded) otherwise.
  auto stageB = [&](int ti, int buf) {
    if (nfull) {
#pragma unroll
      for (int p = 0; p < 8; ++p) {
        const int id  = p * 128 + tid;           // 0..1023 16B chunks
        const int kr  = id >> 5;                 // 0..31
        const int sub = (id & 31) << 2;          // 0..124 step 4
        async_cp16(B + (long long)(ti * BK + kr) * ldb + n0 + sub,
                   &Bfs[buf][kr * BN + sub]);
      }
    } else {
#pragma unroll
      for (int p = 0; p < 8; ++p) {
        const int kr = b_kr + p * 4;
        const float* bp = B + (long long)(ti * BK + kr) * ldb + n0 + b_nq;
        float4 v;
        v.x = (n0 + b_nq + 0 < N) ? bp[0] : 0.f;
        v.y = (n0 + b_nq + 1 < N) ? bp[1] : 0.f;
        v.z = (n0 + b_nq + 2 < N) ? bp[2] : 0.f;
        v.w = (n0 + b_nq + 3 < N) ? bp[3] : 0.f;
        *(float4*)&Bfs[buf][kr * BN + b_nq] = v;
      }
    }
  };

  stageB(0, 0);                                   // prologue: tile 0 in flight

  for (int ti = 0; ti < nk; ++ti) {
    const int buf = ti & 1;
    if (ti + 1 < nk) stageB(ti + 1, buf ^ 1);     // overlap next weight tile
    const int kk = ti * BK;

    // ---- stage+convert A (L2-resident activations), overlaps async B ----
#pragma unroll
    for (int p = 0; p < 8; ++p) {
      const int row = a_r + p * 16;
      const float4 v = *(const float4*)(A + (long long)(m0 + row) * lda + kk + a_kq);
      unsigned short* d = &As[row * SA + a_kq];
      d[0] = f2bf(v.x); d[1] = f2bf(v.y); d[2] = f2bf(v.z); d[3] = f2bf(v.w);
    }

    // Drain this tile's async DMAs (next tile may stay in flight: <= 8/wave).
    if (nfull) {
      if (ti + 1 < nk) __builtin_amdgcn_s_wait_asynccnt(8);
      else             __builtin_amdgcn_s_wait_asynccnt(0);
    }
    __syncthreads();

    // ---- convert B f32 -> bf16, transposed into Bs[n][k] ----
#pragma unroll
    for (int p = 0; p < 8; ++p) {
      const int krow = b_kr + p * 4;
      const float4 v = *(const float4*)&Bfs[buf][krow * BN + b_nq];
      Bs[(b_nq + 0) * SB + krow] = f2bf(v.x);
      Bs[(b_nq + 1) * SB + krow] = f2bf(v.y);
      Bs[(b_nq + 2) * SB + krow] = f2bf(v.z);
      Bs[(b_nq + 3) * SB + krow] = f2bf(v.w);
    }
    __syncthreads();

    // ---- fragments per ISA 16-bit layouts + 16 WMMA ----
    v16bf af[4], bfr[4];
#pragma unroll
    for (int i = 0; i < 4; ++i) {
      const unsigned short* ap = &As[(wm0 + i * 16 + lm) * SA + lh * 8];
      af[i] = ld_frag(ap, ap + 16);
    }
#pragma unroll
    for (int j = 0; j < 4; ++j) {
      const unsigned short* bp = &Bs[(wn0 + j * 16 + lm) * SB + lh * 8];
      bfr[j] = ld_frag(bp, bp + 16);
    }
#pragma unroll
    for (int i = 0; i < 4; ++i)
#pragma unroll
      for (int j = 0; j < 4; ++j)
        acc[i][j] = __builtin_amdgcn_wmma_f32_16x16x32_bf16(
            false, af[i], false, bfr[j], (short)0, acc[i][j], false, false);
    __syncthreads();
  }
#else
  // ------------------- synchronous fallback path -------------------
  for (int kk = 0; kk < K; kk += BK) {
    if (kk + BK < K) {
      __builtin_prefetch((const void*)(A + (long long)(m0 + a_r) * lda + kk + BK), 0, 1);
      __builtin_prefetch((const void*)(B + (long long)(kk + BK + b_kr) * ldb + n0 + b_nq), 0, 1);
    }
#pragma unroll
    for (int p = 0; p < 8; ++p) {
      const int row = a_r + p * 16;
      const float4 v = *(const float4*)(A + (long long)(m0 + row) * lda + kk + a_kq);
      unsigned short* d = &As[row * SA + a_kq];
      d[0] = f2bf(v.x); d[1] = f2bf(v.y); d[2] = f2bf(v.z); d[3] = f2bf(v.w);
    }
#pragma unroll
    for (int p = 0; p < 8; ++p) {
      const int krow = b_kr + p * 4;
      const float* bp = B + (long long)(kk + krow) * ldb + n0 + b_nq;
      float4 v;
      if (n0 + b_nq + 3 < N) {
        v = *(const float4*)bp;
      } else {
        v.x = (n0 + b_nq + 0 < N) ? bp[0] : 0.f;
        v.y = (n0 + b_nq + 1 < N) ? bp[1] : 0.f;
        v.z = (n0 + b_nq + 2 < N) ? bp[2] : 0.f;
        v.w = (n0 + b_nq + 3 < N) ? bp[3] : 0.f;
      }
      Bs[(b_nq + 0) * SB + krow] = f2bf(v.x);
      Bs[(b_nq + 1) * SB + krow] = f2bf(v.y);
      Bs[(b_nq + 2) * SB + krow] = f2bf(v.z);
      Bs[(b_nq + 3) * SB + krow] = f2bf(v.w);
    }
    __syncthreads();
    v16bf af[4], bfr[4];
#pragma unroll
    for (int i = 0; i < 4; ++i) {
      const unsigned short* ap = &As[(wm0 + i * 16 + lm) * SA + lh * 8];
      af[i] = ld_frag(ap, ap + 16);
    }
#pragma unroll
    for (int j = 0; j < 4; ++j) {
      const unsigned short* bp = &Bs[(wn0 + j * 16 + lm) * SB + lh * 8];
      bfr[j] = ld_frag(bp, bp + 16);
    }
#pragma unroll
    for (int i = 0; i < 4; ++i)
#pragma unroll
      for (int j = 0; j < 4; ++j)
        acc[i][j] = __builtin_amdgcn_wmma_f32_16x16x32_bf16(
            false, af[i], false, bfr[j], (short)0, acc[i][j], false, false);
    __syncthreads();
  }
#endif

  // ---- epilogue: C/D layout — lane holds col n=lm, rows r + 8*lh ----
#pragma unroll
  for (int j = 0; j < 4; ++j) {
    const int n = n0 + wn0 + j * 16 + lm;
    if (n >= N) continue;
    const float bv = (flags & EPI_BIAS) ? bias[n] : 0.f;
#pragma unroll
    for (int i = 0; i < 4; ++i) {
#pragma unroll
      for (int r = 0; r < 8; ++r) {
        const int m = m0 + wm0 + i * 16 + r + lh * 8;
        float v = acc[i][j][r] * alpha;
        if (flags & EPI_BIAS)  v += bv;
        if (flags & EPI_GELU)  v = gelu_exact(v);
        if (flags & EPI_MASK)  v *= rowscale[m];
        if (flags & EPI_RESID) v += R[(long long)m * ldc + n];
        C[(long long)m * ldc + n] = v;
      }
    }
  }
}

// ---------------------------------------------------------------------------
// Elementwise / reduction kernels
// ---------------------------------------------------------------------------

// patchify latents (1,4,16,32,32) -> tokens (1024,64); also mask byte -> f32
__global__ __launch_bounds__(256)
void k_patchify(const float* __restrict__ lat, const unsigned char* __restrict__ msk,
                float* __restrict__ tok, float* __restrict__ maskf)
{
  const int id = blockIdx.x * 256 + threadIdx.x;     // 65536
  const int n = id >> 6, k = id & 63;
  const int t = n >> 8, hh = (n >> 4) & 15, ww = n & 15;
  const int c = k >> 2, dy = (k >> 1) & 1, dx = k & 1;
  tok[id] = lat[(((t * 16 + c) * 32) + (hh * 2 + dy)) * 32 + (ww * 2 + dx)];
  if (id < N_TOKENS) maskf[id] = msk[id] ? 1.f : 0.f;
}

// x = [tokens | flag] @ W (65x1024) + b + pos_emb
__global__ __launch_bounds__(256)
void k_inproj(const float* __restrict__ tok, const float* __restrict__ maskf,
              const float* __restrict__ W, const float* __restrict__ bvec,
              const float* __restrict__ et, const float* __restrict__ eh,
              const float* __restrict__ ew, float* __restrict__ X)
{
  const int id = blockIdx.x * 256 + threadIdx.x;     // 1M
  const int n = id >> 10, d = id & 1023;
  const float* tp = tok + n * 64;
  float acc = bvec[d];
#pragma unroll 8
  for (int k = 0; k < 64; ++k) acc += tp[k] * W[k * 1024 + d];
  acc += maskf[n] * W[64 * 1024 + d];
  const int t = n >> 8, hh = (n >> 4) & 15, ww = n & 15;
  acc += et[t * 1024 + d] + eh[hh * 1024 + d] + ew[ww * 1024 + d];
  X[id] = acc;
}

// LayerNorm over 1024 dims; one block (256 threads) per token.
__global__ __launch_bounds__(256)
void k_ln(const float* __restrict__ x, const float* __restrict__ w,
          const float* __restrict__ b, float* __restrict__ y)
{
  __shared__ float s1[256], s2[256];
  const int n = blockIdx.x, t = threadIdx.x;
  const float* xp = x + (long long)n * 1024;
  const float v0 = xp[t], v1 = xp[t + 256], v2 = xp[t + 512], v3 = xp[t + 768];
  s1[t] = v0 + v1 + v2 + v3;
  s2[t] = v0 * v0 + v1 * v1 + v2 * v2 + v3 * v3;
  __syncthreads();
  for (int o = 128; o > 0; o >>= 1) {
    if (t < o) { s1[t] += s1[t + o]; s2[t] += s2[t + o]; }
    __syncthreads();
  }
  const float mean = s1[0] * (1.f / 1024.f);
  const float var  = s2[0] * (1.f / 1024.f) - mean * mean;
  const float rs   = rsqrtf(var + 1e-5f);
  float* yp = y + (long long)n * 1024;
  yp[t]       = (v0 - mean) * rs * w[t]       + b[t];
  yp[t + 256] = (v1 - mean) * rs * w[t + 256] + b[t + 256];
  yp[t + 512] = (v2 - mean) * rs * w[t + 512] + b[t + 512];
  yp[t + 768] = (v3 - mean) * rs * w[t + 768] + b[t + 768];
}

// Local 3x3x3 attention; one thread per (token, head). Zero-padded
// neighbors contribute score 0 to the softmax (matches jnp.pad semantics).
__global__ __launch_bounds__(128)
void k_local_attn(const float* __restrict__ qkv, float* __restrict__ out)
{
  const int id = blockIdx.x * 128 + threadIdx.x;      // 16384
  const int n = id >> 4, h = id & 15;
  const int t = n >> 8, hh = (n >> 4) & 15, ww = n & 15;
  const float* qp = qkv + (long long)n * 3072 + h * 64;
  float q[64];
#pragma unroll
  for (int j = 0; j < 64; ++j) q[j] = qp[j];

  float sc[27];
  for (int nb = 0; nb < 27; ++nb) {
    const int tt = t + nb / 9 - 1;
    const int th = hh + (nb / 3) % 3 - 1;
    const int tw = ww + nb % 3 - 1;
    float s = 0.f;
    if (tt >= 0 && tt < 4 && th >= 0 && th < 16 && tw >= 0 && tw < 16) {
      const int nn = tt * 256 + th * 16 + tw;
      const float* kp = qkv + (long long)nn * 3072 + 1024 + h * 64;
#pragma unroll
      for (int j = 0; j < 64; ++j) s += q[j] * kp[j];
      s *= 0.125f;
    }
    sc[nb] = s;
  }
  float mx = sc[0];
  for (int nb = 1; nb < 27; ++nb) mx = fmaxf(mx, sc[nb]);
  float den = 0.f;
  for (int nb = 0; nb < 27; ++nb) { sc[nb] = __expf(sc[nb] - mx); den += sc[nb]; }
  const float inv = 1.f / den;

  float o[64];
#pragma unroll
  for (int j = 0; j < 64; ++j) o[j] = 0.f;
  for (int nb = 0; nb < 27; ++nb) {
    const int tt = t + nb / 9 - 1;
    const int th = hh + (nb / 3) % 3 - 1;
    const int tw = ww + nb % 3 - 1;
    if (tt >= 0 && tt < 4 && th >= 0 && th < 16 && tw >= 0 && tw < 16) {
      const int nn = tt * 256 + th * 16 + tw;
      const float* vp = qkv + (long long)nn * 3072 + 2048 + h * 64;
      const float p = sc[nb];
#pragma unroll
      for (int j = 0; j < 64; ++j) o[j] += p * vp[j];
    }
  }
  float* op = out + (long long)n * 1024 + h * 64;
#pragma unroll
  for (int j = 0; j < 64; ++j) op[j] = o[j] * inv;
}

// KT[h][d][n] = qkv[n][1024 + h*64 + d]
__global__ __launch_bounds__(256)
void k_transpose_k(const float* __restrict__ qkv, float* __restrict__ kt)
{
  const int id = blockIdx.x * 256 + threadIdx.x;      // 1M
  const int h = id >> 16;
  const int d = (id >> 10) & 63;
  const int n = id & 1023;
  kt[id] = qkv[(long long)n * 3072 + 1024 + h * 64 + d];
}

// Row softmax over 1024 entries; one block (256 threads) per row.
__global__ __launch_bounds__(256)
void k_softmax(float* S)
{
  __shared__ float red[256];
  float* p = S + (long long)blockIdx.x * 1024;
  const int t = threadIdx.x;
  float v0 = p[t], v1 = p[t + 256], v2 = p[t + 512], v3 = p[t + 768];
  red[t] = fmaxf(fmaxf(v0, v1), fmaxf(v2, v3));
  __syncthreads();
  for (int o = 128; o > 0; o >>= 1) {
    if (t < o) red[t] = fmaxf(red[t], red[t + o]);
    __syncthreads();
  }
  const float m = red[0];
  __syncthreads();
  v0 = __expf(v0 - m); v1 = __expf(v1 - m); v2 = __expf(v2 - m); v3 = __expf(v3 - m);
  red[t] = v0 + v1 + v2 + v3;
  __syncthreads();
  for (int o = 128; o > 0; o >>= 1) {
    if (t < o) red[t] += red[t + o];
    __syncthreads();
  }
  const float inv = 1.f / red[0];
  p[t] = v0 * inv; p[t + 256] = v1 * inv; p[t + 512] = v2 * inv; p[t + 768] = v3 * inv;
}

// out (1,4,16,32,32) = unpatchify(tokens + delta)
__global__ __launch_bounds__(256)
void k_unpatch(const float* __restrict__ tok, const float* __restrict__ delta,
               float* __restrict__ out)
{
  const int id = blockIdx.x * 256 + threadIdx.x;      // 65536
  const int t = id >> 14;
  const int c = (id >> 10) & 15;
  const int y = (id >> 5) & 31;
  const int x = id & 31;
  const int n = t * 256 + (y >> 1) * 16 + (x >> 1);
  const int k = c * 4 + (y & 1) * 2 + (x & 1);
  out[id] = tok[n * 64 + k] + delta[n * 64 + k];
}

// ---------------------------------------------------------------------------
// Host orchestration
// ---------------------------------------------------------------------------
static void gemm(hipStream_t st, const float* A, const float* B, float* C,
                 int M, int N, int K, int lda, int ldb, int ldc, int Z,
                 long long sA, long long sB, long long sC,
                 const float* bias, const float* resid, const float* rowscale,
                 float alpha, int flags)
{
  dim3 g((N + BN - 1) / BN, M / BM, Z);
  k_gemm<<<g, 128, 0, st>>>(A, B, C, M, N, K, lda, ldb, ldc, sA, sB, sC,
                            bias, resid, rowscale, alpha, flags);
}

extern "C" void kernel_launch(void* const* d_in, const int* in_sizes, int n_in,
                              void* d_out, int out_size, void* d_ws, size_t ws_size,
                              hipStream_t stream)
{
  const float*         latents = (const float*)d_in[0];
  const unsigned char* tmask   = (const unsigned char*)d_in[1];
  const float* inw   = (const float*)d_in[2];
  const float* inb   = (const float*)d_in[3];
  const float* emb_t = (const float*)d_in[4];
  const float* emb_h = (const float*)d_in[5];
  const float* emb_w = (const float*)d_in[6];
  const float* ln_lw = (const float*)d_in[7];
  const float* ln_lb = (const float*)d_in[8];
  const float* ln_gw = (const float*)d_in[9];
  const float* ln_gb = (const float*)d_in[10];
  const float* ln_fw = (const float*)d_in[11];
  const float* ln_fb = (const float*)d_in[12];
  const float* lqkv  = (const float*)d_in[13];
  const float* lout  = (const float*)d_in[14];
  const float* gqkv  = (const float*)d_in[15];
  const float* gout  = (const float*)d_in[16];
  const float* fw1   = (const float*)d_in[17];
  const float* fb1   = (const float*)d_in[18];
  const float* fw2   = (const float*)d_in[19];
  const float* fb2   = (const float*)d_in[20];
  const float* flnw  = (const float*)d_in[21];
  const float* flnb  = (const float*)d_in[22];
  const float* opw   = (const float*)d_in[23];
  const float* opb   = (const float*)d_in[24];

  // Workspace layout (floats). ~110 MB total.
  float* X     = (float*)d_ws;
  float* LN    = X    + 1048576;
  float* QKV   = LN   + 1048576;
  float* ATT   = QKV  + 3145728;
  float* KT    = ATT  + 1048576;
  float* S     = KT   + 1048576;
  float* H1    = S    + 16777216;
  float* TOK   = H1   + 4194304;
  float* DELTA = TOK  + 65536;
  float* MASKF = DELTA + 65536;

  k_patchify<<<256, 256, 0, stream>>>(latents, tmask, TOK, MASKF);
  k_inproj<<<4096, 256, 0, stream>>>(TOK, MASKF, inw, inb, emb_t, emb_h, emb_w, X);

  for (int l = 0; l < N_LAYERS; ++l) {
    const long long lD   = (long long)l * HIDDEN;
    const long long lDD  = (long long)l * HIDDEN * HIDDEN;
    const long long lD3  = (long long)l * HIDDEN * 3 * HIDDEN;
    const long long lD4  = (long long)l * HIDDEN * 4 * HIDDEN;
    const long long l4D  = (long long)l * 4 * HIDDEN;

    // ---- local attention block ----
    k_ln<<<1024, 256, 0, stream>>>(X, ln_lw + lD, ln_lb + lD, LN);
    gemm(stream, LN, lqkv + lD3, QKV, 1024, 3072, 1024, 1024, 3072, 3072,
         1, 0, 0, 0, nullptr, nullptr, nullptr, 1.f, 0);
    k_local_attn<<<128, 128, 0, stream>>>(QKV, ATT);
    gemm(stream, ATT, lout + lDD, X, 1024, 1024, 1024, 1024, 1024, 1024,
         1, 0, 0, 0, nullptr, X, nullptr, 1.f, EPI_RESID);

    // ---- global attention block ----
    k_ln<<<1024, 256, 0, stream>>>(X, ln_gw + lD, ln_gb + lD, LN);
    gemm(stream, LN, gqkv + lD3, QKV, 1024, 3072, 1024, 1024, 3072, 3072,
         1, 0, 0, 0, nullptr, nullptr, nullptr, 1.f, 0);
    k_transpose_k<<<4096, 256, 0, stream>>>(QKV, KT);
    // S[h] = (Q_h @ K_h^T) / 8 : batched over 16 heads
    gemm(stream, QKV, KT, S, 1024, 1024, 64, 3072, 1024, 1024,
         16, 64, 65536, 1048576, nullptr, nullptr, nullptr, 0.125f, 0);
    k_softmax<<<16384, 256, 0, stream>>>(S);
    // ATT[:, h*64:] = P_h @ V_h
    gemm(stream, S, QKV + 2048, ATT, 1024, 64, 1024, 1024, 3072, 1024,
         16, 1048576, 64, 64, nullptr, nullptr, nullptr, 1.f, 0);
    gemm(stream, ATT, gout + lDD, X, 1024, 1024, 1024, 1024, 1024, 1024,
         1, 0, 0, 0, nullptr, X, MASKF, 1.f, EPI_RESID | EPI_MASK);

    // ---- FFN block ----
    k_ln<<<1024, 256, 0, stream>>>(X, ln_fw + lD, ln_fb + lD, LN);
    gemm(stream, LN, fw1 + lD4, H1, 1024, 4096, 1024, 1024, 4096, 4096,
         1, 0, 0, 0, fb1 + l4D, nullptr, nullptr, 1.f, EPI_BIAS | EPI_GELU);
    gemm(stream, H1, fw2 + lD4, X, 1024, 1024, 4096, 4096, 1024, 1024,
         1, 0, 0, 0, fb2 + lD, X, nullptr, 1.f, EPI_BIAS | EPI_RESID);
  }

  k_ln<<<1024, 256, 0, stream>>>(X, flnw, flnb, LN);
  gemm(stream, LN, opw, DELTA, 1024, 64, 1024, 1024, 64, 64,
       1, 0, 0, 0, opb, nullptr, nullptr, 1.f, EPI_BIAS);
  k_unpatch<<<256, 256, 0, stream>>>(TOK, DELTA, (float*)d_out);
}
